// LSTMClassifier_53669911330986
// MI455X (gfx1250) — matrix operand order repaired and checked by reading, not compiled, not run on hardware
//
#include <hip/hip_runtime.h>
#include <stdint.h>

// ---------------- problem constants ----------------
#define NBATCH 32768
#define NIN    1024
#define NEMB   256
#define NH     256

#define THREADS 512            // 16 waves of 32
#define ROWS_PER_BLOCK 256     // 16 rows per wave
#define NBLOCKS (NBATCH / ROWS_PER_BLOCK)   // 128

// A-matrix LDS: 256 rows x [emb(256) | hidden(256)] bf16, padded pitch
#define ALDS_PITCH_U 260                       // uints per row (1040 B, 16B aligned, bank-skewed)
#define ALDS_BYTES   (ROWS_PER_BLOCK * ALDS_PITCH_U * 4)   // 266,240
// weight strip buffer:
//   phase-1: two 32-wide K slices of Wemb: 2 x (256 rows x 20 uints) = 40,960 B
//   phase-3: 2 gates x 16 rows x 260 uints = 33,280 B
#define P1_SLICE_U   (256 * 20)                            // uints per phase-1 slice
#define STRIP_BYTES  (2 * P1_SLICE_U * 4)                  // 40,960
#define SMEM_BYTES   (ALDS_BYTES + STRIP_BYTES)            // 307,200 <= 320KB

typedef __attribute__((ext_vector_type(16))) __bf16 bf16x16;
typedef __attribute__((ext_vector_type(8)))  float  floatx8;

union FragU { unsigned u[8]; bf16x16 v; };

__device__ __forceinline__ unsigned pack2bf(float a, float b) {
  unsigned ua = __float_as_uint(a), ub = __float_as_uint(b);
  ua = (ua + 0x7FFFu + ((ua >> 16) & 1u)) >> 16;          // RNE
  ub = (ub + 0x7FFFu + ((ub >> 16) & 1u)) >> 16;
  return (ub << 16) | (ua & 0xFFFFu);
}
__device__ __forceinline__ unsigned short bf1(float a) {
  unsigned ua = __float_as_uint(a);
  return (unsigned short)((ua + 0x7FFFu + ((ua >> 16) & 1u)) >> 16);
}

__device__ __forceinline__ floatx8 wmma_bf16(bf16x16 a, bf16x16 b, floatx8 c) {
  return __builtin_amdgcn_wmma_f32_16x16x32_bf16(false, a, false, b, (short)0, c, false, false);
}

// A-frag (16x32 bf16) straight from global f32; rowPtr = start of this lane's row.
// ISA A layout: lane half hl: k = k0 + hl*8 + {0,2,4,6} (u0..u3) and +16 (u4..u7); pairs contiguous.
__device__ __forceinline__ bf16x16 aFragGlobal(const float* rowPtr, int k0, int hl) {
  const float* p = rowPtr + k0 + hl * 8;
  FragU f;
#pragma unroll
  for (int v = 0; v < 8; ++v) {
    int k = (v < 4) ? 2 * v : 16 + 2 * (v - 4);
    float2 t = *(const float2*)(p + k);
    f.u[v] = pack2bf(t.x, t.y);
  }
  return f.v;
}

// A-frag from packed-bf16 LDS row (uints). k0u = k0/2.
__device__ __forceinline__ bf16x16 aFragLds(const uint32_t* rowU, int k0u, int hl) {
  const uint32_t* p = rowU + k0u + hl * 4;
  uint4 q0 = *(const uint4*)(p);
  uint4 q1 = *(const uint4*)(p + 8);
  FragU f;
  f.u[0]=q0.x; f.u[1]=q0.y; f.u[2]=q0.z; f.u[3]=q0.w;
  f.u[4]=q1.x; f.u[5]=q1.y; f.u[6]=q1.z; f.u[7]=q1.w;
  return f.v;
}

// B-frag (32x16 bf16) from packed-bf16 LDS row (lane = out column n).
// ISA B layout: lane half hl holds K = k0 + hl*8 + {0..7} (u0..u3) and +16 (u4..u7), same as A.
__device__ __forceinline__ bf16x16 bFragLds(const uint32_t* rowU, int k0u, int hl) {
  const uint32_t* p = rowU + k0u + hl * 4;
  uint4 q0 = *(const uint4*)(p);
  uint4 q1 = *(const uint4*)(p + 8);
  FragU f;
  f.u[0]=q0.x; f.u[1]=q0.y; f.u[2]=q0.z; f.u[3]=q0.w;
  f.u[4]=q1.x; f.u[5]=q1.y; f.u[6]=q1.z; f.u[7]=q1.w;
  return f.v;
}

struct Params {
  const float *input, *hidden, *cell;
  const float *Wemb, *bemb;
  const float *Wg0, *Wg1, *Wg2, *Wg3;        // f, i, c, o  (EMB contribution)
  const float *bg0, *bg1, *bg2, *bg3;
  const float *Wh0, *Wh1, *Wh2, *Wh3;        // hidden contribution
  const float *bh0, *bh1, *bh2, *bh3;
  const float *Wend, *bend;
  float *logp, *newh, *newc;
};

__global__ __launch_bounds__(THREADS, 1)
void lstm_fused(Params P) {
  extern __shared__ char smem[];
  uint32_t* alds  = (uint32_t*)smem;                      // 256 x 260 uints
  uint32_t* strip = (uint32_t*)(smem + ALDS_BYTES);

  const int tid  = threadIdx.x;
  const int wave = tid >> 5;
  const int lane = tid & 31;
  const int lq   = lane & 15;                             // M row (A) / N col (B,C)
  const int hl   = lane >> 4;
  const int rowBase = blockIdx.x * ROWS_PER_BLOCK;
  const int waveRow = wave * 16;                          // block-local first row of this wave

  // ---------- Phase 0: stage hidden (f32 -> bf16) into alds columns [256..511] ----------
  {
    int r   = tid >> 1;                                   // 0..255
    int seg = tid & 1;                                    // which 64-uint half of the row
    const float* hrow = P.hidden + (size_t)(rowBase + r) * NH + seg * 128;
    uint32_t* dst = alds + (size_t)r * ALDS_PITCH_U + 128 + seg * 64;
#pragma unroll
    for (int i = 0; i < 16; ++i) {
      float4 f0 = ((const float4*)hrow)[2 * i];
      float4 f1 = ((const float4*)hrow)[2 * i + 1];
      uint4 o = make_uint4(pack2bf(f0.x, f0.y), pack2bf(f0.z, f0.w),
                           pack2bf(f1.x, f1.y), pack2bf(f1.z, f1.w));
      ((uint4*)dst)[i] = o;
    }
  }

  // ---------- Phase 1: embedded = input @ Wemb^T  (K = 1024, bf16 WMMA, f32 accum) ----------
  floatx8 zf = {};
  floatx8 acc[16];
#pragma unroll
  for (int i = 0; i < 16; ++i) acc[i] = zf;

  const float* aRow = P.input + (size_t)(rowBase + waveRow + lq) * NIN;

  for (int ks2 = 0; ks2 < NIN / 64; ++ks2) {              // 16 staging rounds, 64 K each
    __syncthreads();
    { // stage Wemb[:, ks2*64 .. +64) as packed bf16: 2 slices x (256 rows x 16 uints, pitch 20)
      int s = tid >> 8;                                   // sub-slice 0/1
      int r = tid & 255;                                  // row
      const float* w = P.Wemb + (size_t)r * NIN + ks2 * 64 + s * 32;
      uint32_t* dst = strip + s * P1_SLICE_U + r * 20;
#pragma unroll
      for (int i = 0; i < 4; ++i) {
        float4 fA = ((const float4*)w)[2 * i];
        float4 fB = ((const float4*)w)[2 * i + 1];
        ((uint4*)dst)[i] = make_uint4(pack2bf(fA.x, fA.y), pack2bf(fA.z, fA.w),
                                      pack2bf(fB.x, fB.y), pack2bf(fB.z, fB.w));
      }
    }
    __syncthreads();

    if (ks2 + 1 < NIN / 64) __builtin_prefetch(aRow + (ks2 + 1) * 64, 0, 3);
    // issue both A-fragments up front so sub-slice 1's global loads overlap slice 0's WMMAs
    bf16x16 aF0 = aFragGlobal(aRow, ks2 * 64, hl);
    bf16x16 aF1 = aFragGlobal(aRow, ks2 * 64 + 32, hl);

#pragma unroll
    for (int s = 0; s < 2; ++s) {
      bf16x16 aF = (s == 0) ? aF0 : aF1;
      const uint32_t* sbase = strip + s * P1_SLICE_U;
      // software-pipeline B-fragments one n-tile ahead
      bf16x16 bCur = bFragLds(sbase + (size_t)lq * 20, 0, hl);
#pragma unroll
      for (int nt = 0; nt < 16; ++nt) {
        bf16x16 bNxt = bCur;
        if (nt < 15) bNxt = bFragLds(sbase + (size_t)((nt + 1) * 16 + lq) * 20, 0, hl);
        acc[nt] = wmma_bf16(aF, bCur, acc[nt]);
        bCur = bNxt;
      }
    }
  }

  // bias + store embedded tile as bf16 into alds columns [0..255] (wave-local rows)
  {
    int mloc = waveRow + hl * 8;
#pragma unroll
    for (int nt = 0; nt < 16; ++nt) {
      int n = nt * 16 + lq;
      float bias = P.bemb[n];
#pragma unroll
      for (int v = 0; v < 8; ++v) {
        *(unsigned short*)(smem + (size_t)(mloc + v) * (ALDS_PITCH_U * 4) + n * 2) =
            bf1(acc[nt][v] + bias);
      }
    }
  }

  // ---------- Phase 2/3: gates (K=512 over [emb|hid]) + LSTM elementwise + logits ----------
  float l0[8], l1[8];
#pragma unroll
  for (int v = 0; v < 8; ++v) { l0[v] = 0.f; l1[v] = 0.f; }

  const uint32_t* aldsRow = alds + (size_t)(waveRow + lq) * ALDS_PITCH_U;

  for (int nt = 0; nt < 16; ++nt) {
    floatx8 gaF = zf, gaI = zf, gaC = zf, gaO = zf;

#pragma unroll
    for (int pass = 0; pass < 2; ++pass) {                // pass0: f,i   pass1: c,o
      __syncthreads();
      { // stage [W_g | W_hg] strips for 2 gates: 2 x 16 rows x 256 uints (pitch 260)
        int g   = tid >> 8;                               // 0 or 1 within the pass
        int r   = (tid >> 4) & 15;
        int seg = tid & 15;                               // 16 uints = 32 k-values
        const float* Wa = (pass == 0) ? (g == 0 ? P.Wg0 : P.Wg1) : (g == 0 ? P.Wg2 : P.Wg3);
        const float* Wb = (pass == 0) ? (g == 0 ? P.Wh0 : P.Wh1) : (g == 0 ? P.Wh2 : P.Wh3);
        int kloc = seg * 32;
        const float* src = (kloc < NEMB)
                             ? (Wa + (size_t)(nt * 16 + r) * NEMB + kloc)
                             : (Wb + (size_t)(nt * 16 + r) * NH + (kloc - NEMB));
        uint32_t* dst = strip + (size_t)(g * 16 + r) * ALDS_PITCH_U + seg * 16;
#pragma unroll
        for (int i = 0; i < 4; ++i) {
          float4 fA = ((const float4*)src)[2 * i];
          float4 fB = ((const float4*)src)[2 * i + 1];
          ((uint4*)dst)[i] = make_uint4(pack2bf(fA.x, fA.y), pack2bf(fA.z, fA.w),
                                        pack2bf(fB.x, fB.y), pack2bf(fB.z, fB.w));
        }
      }
      __syncthreads();

      const uint32_t* s0 = strip + (size_t)lq * ALDS_PITCH_U;
      const uint32_t* s1 = strip + (size_t)(16 + lq) * ALDS_PITCH_U;
      // software-pipeline A + both B fragments one k-step ahead
      bf16x16 aC  = aFragLds(aldsRow, 0, hl);
      bf16x16 b0C = bFragLds(s0, 0, hl);
      bf16x16 b1C = bFragLds(s1, 0, hl);
#pragma unroll
      for (int ks = 0; ks < 16; ++ks) {                   // K = 512
        bf16x16 aN = aC, b0N = b0C, b1N = b1C;
        if (ks < 15) {
          aN  = aFragLds(aldsRow, (ks + 1) * 16, hl);
          b0N = bFragLds(s0, (ks + 1) * 16, hl);
          b1N = bFragLds(s1, (ks + 1) * 16, hl);
        }
        if (pass == 0) { gaF = wmma_bf16(aC, b0C, gaF); gaI = wmma_bf16(aC, b1C, gaI); }
        else           { gaC = wmma_bf16(aC, b0C, gaC); gaO = wmma_bf16(aC, b1C, gaO); }
        aC = aN; b0C = b0N; b1C = b1N;
      }
    }

    // elementwise LSTM update for this 16-column tile
    int n = nt * 16 + lq;
    float bF = P.bg0[n] + P.bh0[n];
    float bI = P.bg1[n] + P.bh1[n];
    float bC = P.bg2[n] + P.bh2[n];
    float bO = P.bg3[n] + P.bh3[n];
    float w0 = P.Wend[n], w1 = P.Wend[NH + n];
    int mglob = rowBase + waveRow + hl * 8;
#pragma unroll
    for (int v = 0; v < 8; ++v) {
      size_t idx = (size_t)(mglob + v) * NH + n;
      float fg = 1.f / (1.f + __expf(-(gaF[v] + bF)));
      float ig = 1.f / (1.f + __expf(-(gaI[v] + bI)));
      float ct = tanhf(gaC[v] + bC);
      float og = 1.f / (1.f + __expf(-(gaO[v] + bO)));
      float nc = fg * P.cell[idx] + ig * ct;
      float nh = og * tanhf(nc);
      P.newc[idx] = nc;
      P.newh[idx] = nh;
      l0[v] += nh * w0;
      l1[v] += nh * w1;
    }
  }

  // ---------- logits: reduce over the 16-lane N dimension, then 2-class log_softmax ----------
  float be0 = P.bend[0], be1 = P.bend[1];
#pragma unroll
  for (int v = 0; v < 8; ++v) {
    float a = l0[v], b = l1[v];
#pragma unroll
    for (int off = 1; off < 16; off <<= 1) {
      a += __shfl_xor(a, off, 32);
      b += __shfl_xor(b, off, 32);
    }
    if (lq == 0) {
      a += be0; b += be1;
      float mx  = fmaxf(a, b);
      float lse = mx + __logf(__expf(a - mx) + __expf(b - mx));
      int m = rowBase + waveRow + hl * 8 + v;
      *(float2*)(P.logp + (size_t)2 * m) = make_float2(a - lse, b - lse);
    }
  }
}

extern "C" void kernel_launch(void* const* d_in, const int* in_sizes, int n_in,
                              void* d_out, int out_size, void* d_ws, size_t ws_size,
                              hipStream_t stream) {
  (void)in_sizes; (void)n_in; (void)out_size; (void)d_ws; (void)ws_size;
  Params P;
  P.input  = (const float*)d_in[0];
  P.hidden = (const float*)d_in[1];
  P.cell   = (const float*)d_in[2];
  P.Wemb   = (const float*)d_in[3];  P.bemb = (const float*)d_in[4];
  P.Wg0 = (const float*)d_in[5];   P.bg0 = (const float*)d_in[6];
  P.Wg1 = (const float*)d_in[7];   P.bg1 = (const float*)d_in[8];
  P.Wg2 = (const float*)d_in[9];   P.bg2 = (const float*)d_in[10];
  P.Wg3 = (const float*)d_in[11];  P.bg3 = (const float*)d_in[12];
  P.Wh0 = (const float*)d_in[13];  P.bh0 = (const float*)d_in[14];
  P.Wh1 = (const float*)d_in[15];  P.bh1 = (const float*)d_in[16];
  P.Wh2 = (const float*)d_in[17];  P.bh2 = (const float*)d_in[18];
  P.Wh3 = (const float*)d_in[19];  P.bh3 = (const float*)d_in[20];
  P.Wend = (const float*)d_in[21]; P.bend = (const float*)d_in[22];

  float* out = (float*)d_out;
  P.logp = out;                                            // [B,2]
  P.newh = out + (size_t)NBATCH * 2;                       // [B,256]
  P.newc = out + (size_t)NBATCH * 2 + (size_t)NBATCH * NH; // [B,256]

  hipLaunchKernelGGL(lstm_fused, dim3(NBLOCKS), dim3(THREADS), SMEM_BYTES, stream, P);
}
